// TransformerEncoder_1683627180830
// MI455X (gfx1250) — matrix operand order repaired and checked by reading, not compiled, not run on hardware
//
#include <hip/hip_runtime.h>
#include <hip/hip_bf16.h>

// ---------------------------------------------------------------------------
// CDNA5 (gfx1250) fused transformer encoder block.
// All GEMMs run on v_wmma_f32_16x16x32_bf16 (wave32 WMMA), f32 accumulation.
// ---------------------------------------------------------------------------

typedef __attribute__((ext_vector_type(16))) __bf16 v16bf;
typedef __attribute__((ext_vector_type(8)))  float  v8f;

union FragBF {           // one WMMA 16-wide bf16 operand, viewed 3 ways
  v16bf v;
  uint4 q[2];
  unsigned int d[8];
};

__device__ static inline v8f wmma_bf16(v16bf a, v16bf b, v8f c) {
  // 8 args: (neg_a, A, neg_b, B, c_mod, C, reuse_a, reuse_b)
  return __builtin_amdgcn_wmma_f32_16x16x32_bf16(false, a, false, b,
                                                 (short)0, c, false, false);
}

__device__ static inline float gelu_exact(float x) {
  return 0.5f * x * (1.0f + erff(x * 0.70710678118654752f));
}

// branch-free zeroing: keeps the load unconditional (no exec predication)
__device__ static inline uint4 and4(uint4 a, unsigned m) {
  a.x &= m; a.y &= m; a.z &= m; a.w &= m;
  return a;
}

// ---------------------------------------------------------------------------
// f32 -> bf16 elementwise convert (weights)
// ---------------------------------------------------------------------------
__global__ __launch_bounds__(256) void f2bf_kernel(const float* __restrict__ in,
                                                   __bf16* __restrict__ out, int n) {
  int i = blockIdx.x * 256 + threadIdx.x;
  if (i < n) out[i] = (__bf16)in[i];
}

// ---------------------------------------------------------------------------
// LayerNorm over C=768, one 256-thread block per token, bf16 output.
// ---------------------------------------------------------------------------
__global__ __launch_bounds__(256) void ln_kernel(const float* __restrict__ x,
                                                 const float* __restrict__ g,
                                                 const float* __restrict__ be,
                                                 __bf16* __restrict__ out) {
  __shared__ float red[256];
  const int tok = blockIdx.x;
  const float* xp = x + (size_t)tok * 768;
  const int t = threadIdx.x;

  float s = 0.0f;
#pragma unroll
  for (int p = 0; p < 3; ++p) s += xp[t + p * 256];
  red[t] = s;
  __syncthreads();
  for (int off = 128; off > 0; off >>= 1) {
    if (t < off) red[t] += red[t + off];
    __syncthreads();
  }
  const float mu = red[0] * (1.0f / 768.0f);
  __syncthreads();

  float v = 0.0f;
#pragma unroll
  for (int p = 0; p < 3; ++p) {
    float d = xp[t + p * 256] - mu;
    v += d * d;
  }
  red[t] = v;
  __syncthreads();
  for (int off = 128; off > 0; off >>= 1) {
    if (t < off) red[t] += red[t + off];
    __syncthreads();
  }
  const float rstd = rsqrtf(red[0] * (1.0f / 768.0f) + 1e-5f);

#pragma unroll
  for (int p = 0; p < 3; ++p) {
    int i = t + p * 256;
    out[(size_t)tok * 768 + i] = (__bf16)((xp[i] - mu) * rstd * g[i] + be[i]);
  }
}

// ---------------------------------------------------------------------------
// LDS-staged bf16 WMMA GEMM:  out[M,N] = A[M,K] @ W[K,N]  (+bias,+resid,GELU)
// Block = 256 threads (8 waves). Block tile: 64 (M) x 128 (N). K step: 32.
// Each wave: 4 MxN 16x16 accumulators sharing one B fragment.
// Requires: M % 64 == 0, N % 128 == 0, K % 32 == 0.
// ---------------------------------------------------------------------------
template <bool BIAS, bool RESID, bool GELU_ACT, bool OUT_BF16>
__global__ __launch_bounds__(256) void gemm_wmma(
    const __bf16* __restrict__ A, const __bf16* __restrict__ W,
    const float* __restrict__ bias, const float* __restrict__ resid,
    float* __restrict__ outF, __bf16* __restrict__ outB,
    int M, int N, int K) {
  __shared__ __align__(16) unsigned int As32[64 * 17];  // 64 rows x 32 bf16, +1 dword pad/row
  __shared__ __align__(16) unsigned int Ws32[32 * 64];  // 32 rows x 128 bf16
  const __bf16* Ws = reinterpret_cast<const __bf16*>(Ws32);

  const int t = threadIdx.x;
  const int wave = t >> 5;
  const int lane = t & 31;
  const int hi = lane >> 4;     // 0: lanes 0-15, 1: lanes 16-31
  const int lm = lane & 15;

  const int mbase = blockIdx.x * 64;
  const int nbase = blockIdx.y * 128;

  const unsigned int* A32 = reinterpret_cast<const unsigned int*>(A);
  const uint4* W128 = reinterpret_cast<const uint4*>(W);

  v8f acc[4] = {};
  const int ksteps = K >> 5;

  for (int kk = 0; kk < ksteps; ++kk) {
    // --- stage A tile: 64x32 bf16 = 1024 dwords, 4 per thread (coalesced b32) ---
    {
      int dd = t;
#pragma unroll
      for (int p = 0; p < 4; ++p, dd += 256) {
        int row = dd >> 4;       // 16 dwords per logical row
        int cd = dd & 15;
        As32[row * 17 + cd] = A32[(size_t)(mbase + row) * (K >> 1) + (kk << 4) + cd];
      }
    }
    // --- stage W tile: 32x128 bf16 = 512 uint4, 2 per thread (coalesced b128) ---
    {
      int q = t;
#pragma unroll
      for (int p = 0; p < 2; ++p, q += 256) {
        int row = q >> 4;        // 16 uint4 per row
        int c4 = q & 15;
        reinterpret_cast<uint4*>(Ws32)[q] =
            W128[(size_t)(kk * 32 + row) * (N >> 3) + (nbase >> 3) + c4];
      }
    }
    if (kk + 1 < ksteps) {
      // gfx1250 speculative prefetch of next K-slab (global_prefetch_b8)
      __builtin_prefetch((const void*)&A[(size_t)(mbase + lane) * K + (kk + 1) * 32], 0, 1);
      __builtin_prefetch((const void*)&W[(size_t)((kk + 1) * 32 + (t >> 3)) * N + nbase], 0, 1);
    }
    __syncthreads();

    // --- A fragments: vectorized dword reads, contiguous k order (k = 16*hi + s) ---
    // element pair (2j,2j+1): k = 16*hi + 2j -> dword = j + hi*8
    FragBF a[4];
#pragma unroll
    for (int s2 = 0; s2 < 4; ++s2) {
#pragma unroll
      for (int j = 0; j < 8; ++j) {
        int dw = j + (hi << 3);
        a[s2].d[j] = As32[(s2 * 16 + lm) * 17 + dw];
      }
    }
    // --- B fragment: K split across lane halves, column = wave*16 + lm ---
    v16bf bfrag;
#pragma unroll
    for (int i = 0; i < 16; ++i)
      bfrag[i] = Ws[(i + (hi << 4)) * 128 + (wave << 4) + lm];

#pragma unroll
    for (int s2 = 0; s2 < 4; ++s2)
      acc[s2] = wmma_bf16(a[s2].v, bfrag, acc[s2]);
    __syncthreads();
  }

  // --- epilogue: C/D layout: lane half selects M+8, VGPR r = M row ---
  const int col = nbase + (wave << 4) + lm;
  const float bv = BIAS ? bias[col] : 0.0f;
#pragma unroll
  for (int s2 = 0; s2 < 4; ++s2) {
#pragma unroll
    for (int r = 0; r < 8; ++r) {
      const int row = mbase + s2 * 16 + r + (hi << 3);
      float v0 = acc[s2][r] + bv;
      if (GELU_ACT) v0 = gelu_exact(v0);
      if (RESID) v0 += resid[(size_t)row * N + col];
      if (OUT_BF16) outB[(size_t)row * N + col] = (__bf16)v0;
      else          outF[(size_t)row * N + col] = v0;
    }
  }
}

// ---------------------------------------------------------------------------
// Attention: one wave per (query-tile, head, batch).
// qkv per token: [Q(768) | K(768) | V(768)]; head h -> 64-dim slice.
// Q/K fragments: unconditional clamped uint4 global loads + mask (no exec
// predication). Scores 16x608 in LDS (577 keys, -inf pad), 2-lanes-per-row
// softmax with __shfl_xor combine, V staged in LDS, PV via WMMA.
// ---------------------------------------------------------------------------
__global__ __launch_bounds__(32) void attn_kernel(const __bf16* __restrict__ qkv,
                                                  __bf16* __restrict__ o) {
  __shared__ float  S[16 * 608];
  __shared__ __align__(16) __bf16 P[16 * 608];
  __shared__ __align__(16) __bf16 Vt[32 * 64];

  const int lane = threadIdx.x;
  const int hi = lane >> 4;
  const int lm = lane & 15;
  const int qt = blockIdx.x;   // 0..36
  const int h = blockIdx.y;    // 0..11
  const int b = blockIdx.z;    // 0..63
  const int N = 577;
  const float scale = 0.125f;  // 1/sqrt(64)
  const float NEGINF = -3.0e38f;

  const uint4* qkv4 = reinterpret_cast<const uint4*>(qkv);  // 8 bf16 per uint4

  // init only the pad columns 592..607 (QK loop writes 0..591 unconditionally)
  for (int i = lane; i < 16 * 16; i += 32) {
    int r = i >> 4, c = i & 15;
    S[r * 608 + 592 + c] = NEGINF;
  }

  // ---- Q fragments (A-matrix, 16 rows x 64 dims = 2 k-chunks) ----
  FragBF aq0, aq1;
  {
    const int qrow = qt * 16 + lm;
    const bool qv = qrow < N;
    const unsigned qm = qv ? 0xFFFFFFFFu : 0u;
    const size_t tq = (size_t)(b * N + (qv ? qrow : 0)) * 288 + h * 8;  // uint4 units
    aq0.q[0] = and4(qkv4[tq + hi],     qm);   // d =  0+hi*8 ..  7+hi*8
    aq0.q[1] = and4(qkv4[tq + 2 + hi], qm);   // d = 16+hi*8 .. 23+hi*8
    aq1.q[0] = and4(qkv4[tq + 4 + hi], qm);   // d = 32+hi*8 ..
    aq1.q[1] = and4(qkv4[tq + 6 + hi], qm);   // d = 48+hi*8 ..
  }

  // ---- S = scale * Q @ K^T over 37 key tiles ----
  for (int j = 0; j < 37; ++j) {
    const int kidx = j * 16 + lm;
    const bool kvb = kidx < N;
    const unsigned km = kvb ? 0xFFFFFFFFu : 0u;
    const size_t tk = (size_t)(b * N + (kvb ? kidx : 0)) * 288 + 96 + h * 8;
    FragBF bk0, bk1;
    bk0.q[0] = and4(qkv4[tk + hi],     km);   // d =  0+hi*8 ..  7+hi*8
    bk0.q[1] = and4(qkv4[tk + 2 + hi], km);   // d = 16+hi*8 .. 23+hi*8
    bk1.q[0] = and4(qkv4[tk + 4 + hi], km);   // d = 32+hi*8 ..
    bk1.q[1] = and4(qkv4[tk + 6 + hi], km);

    v8f s = {};
    s = wmma_bf16(aq0.v, bk0.v, s);
    s = wmma_bf16(aq1.v, bk1.v, s);
#pragma unroll
    for (int r = 0; r < 8; ++r) {
      int m = r + (hi << 3);
      S[m * 608 + j * 16 + lm] = kvb ? s[r] * scale : NEGINF;  // select-data store
    }
  }
  __syncthreads();

  // ---- softmax: 2 lanes per row, halves combined via shfl_xor(16) ----
  {
    float* row = &S[lm * 608];
    const int c0 = hi * 304;
    float mx = NEGINF;
    for (int c = c0; c < c0 + 304; ++c) mx = fmaxf(mx, row[c]);
    mx = fmaxf(mx, __shfl_xor(mx, 16, 32));
    float sum = 0.0f;
    for (int c = c0; c < c0 + 304; ++c) {
      float e = __expf(row[c] - mx);
      row[c] = e;
      sum += e;
    }
    sum += __shfl_xor(sum, 16, 32);
    const float inv = 1.0f / sum;
    __bf16* prow = &P[lm * 608];
    for (int c = c0; c < c0 + 304; ++c) prow[c] = (__bf16)(row[c] * inv);
  }
  __syncthreads();

  // ---- O = P(16x608) @ V(608x64): 19 key-chunks of 32, 4 d-tiles ----
  const uint4* P4 = reinterpret_cast<const uint4*>(P);  // 8 bf16 per uint4; row = 76 uint4
  v8f oacc[4] = {};
  for (int kc = 0; kc < 19; ++kc) {
    // stage V chunk: 32 keys x 32 dims = 128 uint4, 4 per lane, coalesced
#pragma unroll
    for (int p = 0; p < 4; ++p) {
      int q = lane + p * 32;       // 0..127
      int row = q >> 2, c4 = q & 3;
      int key = kc * 32 + row;
      bool kv2 = key < N;
      unsigned m2 = kv2 ? 0xFFFFFFFFu : 0u;
      size_t tv = (size_t)(b * N + (kv2 ? key : 0)) * 288 + 192 + h * 8 + c4;
      reinterpret_cast<uint4*>(Vt)[q] = and4(qkv4[tv], m2);
    }
    __syncthreads();

    // A fragment from P: two contiguous 8-element runs per lane (b128 ds reads)
    FragBF ap;
    ap.q[0] = P4[lm * 76 + kc * 4 + 2 * hi];
    ap.q[1] = P4[lm * 76 + kc * 4 + 2 * hi + 1];

#pragma unroll
    for (int dt = 0; dt < 4; ++dt) {
      v16bf bvf;
#pragma unroll
      for (int i = 0; i < 16; ++i)
        bvf[i] = Vt[(i + (hi << 4)) * 64 + dt * 16 + lm];
      oacc[dt] = wmma_bf16(ap.v, bvf, oacc[dt]);
    }
    __syncthreads();
  }

  // ---- write O back to [token][C] bf16 ----
#pragma unroll
  for (int dt = 0; dt < 4; ++dt) {
#pragma unroll
    for (int r = 0; r < 8; ++r) {
      int m = r + (hi << 3);
      int row = qt * 16 + m;
      if (row < N)
        o[(size_t)(b * N + row) * 768 + h * 64 + dt * 16 + lm] = (__bf16)oacc[dt][r];
    }
  }
}

// ---------------------------------------------------------------------------
// Host-side orchestration
// ---------------------------------------------------------------------------
extern "C" void kernel_launch(void* const* d_in, const int* in_sizes, int n_in,
                              void* d_out, int out_size, void* d_ws, size_t ws_size,
                              hipStream_t stream) {
  const float* x      = (const float*)d_in[0];
  const float* ln1_g  = (const float*)d_in[1];
  const float* ln1_b  = (const float*)d_in[2];
  const float* w_qkv  = (const float*)d_in[3];
  const float* w_proj = (const float*)d_in[4];
  const float* b_proj = (const float*)d_in[5];
  const float* ln2_g  = (const float*)d_in[6];
  const float* ln2_b  = (const float*)d_in[7];
  const float* w_fc1  = (const float*)d_in[8];
  const float* b_fc1  = (const float*)d_in[9];
  const float* w_fc2  = (const float*)d_in[10];
  const float* b_fc2  = (const float*)d_in[11];
  float* out = (float*)d_out;

  const int M = 64 * 577;  // 36928 tokens; 36928 % 64 == 0 (577 tiles)

  char* ws = (char*)d_ws;
  size_t off = 0;
  auto carve = [&](size_t bytes) -> char* {
    char* p = ws + off;
    off += (bytes + 255) & ~(size_t)255;
    return p;
  };
  __bf16* hb     = (__bf16*)carve((size_t)M * 768 * 2);     // LN1 out
  __bf16* qkvb   = (__bf16*)carve((size_t)M * 2304 * 2);    // QKV
  __bf16* ob     = (__bf16*)carve((size_t)M * 768 * 2);     // attn out
  float*  x1     = (float*) carve((size_t)M * 768 * 4);     // x + proj
  __bf16* h2b    = (__bf16*)carve((size_t)M * 768 * 2);     // LN2 out
  __bf16* f1b    = (__bf16*)carve((size_t)M * 3072 * 2);    // GELU(fc1)
  __bf16* wqkvb  = (__bf16*)carve((size_t)768 * 2304 * 2);
  __bf16* wprojb = (__bf16*)carve((size_t)768 * 768 * 2);
  __bf16* wfc1b  = (__bf16*)carve((size_t)768 * 3072 * 2);
  __bf16* wfc2b  = (__bf16*)carve((size_t)3072 * 768 * 2);

  // 1) weights -> bf16
  auto cvt = [&](const float* src, __bf16* dst, int n) {
    f2bf_kernel<<<(n + 255) / 256, 256, 0, stream>>>(src, dst, n);
  };
  cvt(w_qkv, wqkvb, 768 * 2304);
  cvt(w_proj, wprojb, 768 * 768);
  cvt(w_fc1, wfc1b, 768 * 3072);
  cvt(w_fc2, wfc2b, 3072 * 768);

  // 2) LN1
  ln_kernel<<<M, 256, 0, stream>>>(x, ln1_g, ln1_b, hb);

  // 3) QKV GEMM (bf16 out)
  gemm_wmma<false, false, false, true><<<dim3(M / 64, 2304 / 128), 256, 0, stream>>>(
      hb, wqkvb, nullptr, nullptr, nullptr, qkvb, M, 2304, 768);

  // 4) attention
  attn_kernel<<<dim3(37, 12, 64), 32, 0, stream>>>(qkvb, ob);

  // 5) proj + bias + residual(x) -> x1 (f32)
  gemm_wmma<true, true, false, false><<<dim3(M / 64, 768 / 128), 256, 0, stream>>>(
      ob, wprojb, b_proj, x, x1, nullptr, M, 768, 768);

  // 6) LN2
  ln_kernel<<<M, 256, 0, stream>>>(x1, ln2_g, ln2_b, h2b);

  // 7) FC1 + bias + exact GELU (bf16 out)
  gemm_wmma<true, false, true, true><<<dim3(M / 64, 3072 / 128), 256, 0, stream>>>(
      h2b, wfc1b, b_fc1, nullptr, nullptr, f1b, M, 3072, 768);

  // 8) FC2 + bias + residual(x1) -> d_out (f32)
  gemm_wmma<true, true, false, false><<<dim3(M / 64, 768 / 128), 256, 0, stream>>>(
      f1b, wfc2b, b_fc2, x1, out, nullptr, M, 768, 3072);
}
